// SeqAddAttentionPooling_43568148251343
// MI455X (gfx1250) — matrix operand order repaired and checked by reading, not compiled, not run on hardware
//
#include <hip/hip_runtime.h>
#include <hip/hip_bf16.h>

// ---------------------------------------------------------------------------
// Additive attention pooling, fused for MI455X (gfx1250, wave32, WMMA).
//   Wc = W + U  (one GEMM instead of two)
//   scores[b,s] = sum_u tanh( (X @ Wc^T)[b,s,u] ) * v[u]   -- H never stored
//   alpha = masked softmax over s ;  out[b,e] = sum_s alpha * X
// GEMM precision: bf16 A/B, fp32 accumulate via v_wmma_f32_16x16x32_bf16.
// 2x2 register tile per wave; an opaque per-nt offset stops LICM from
// hoisting the (nt-invariant) LDS A-operand loads out of the n-tile loop,
// which previously spilled 2 KB/lane of A data to scratch.
// ---------------------------------------------------------------------------

typedef __attribute__((ext_vector_type(16))) __bf16   v16bf;
typedef __attribute__((ext_vector_type(8)))  float    v8f;
typedef __attribute__((ext_vector_type(4)))  unsigned int v4u;

union ABPack { v4u q[2]; v16bf v; };

#define BATCH   32
#define SEQ     2048
#define EDIM    1024
#define UDIM    1024
#define MTILE   64          // rows of X per workgroup
#define LSTRIDE 1032        // bf16 elems per LDS row (pad 8 -> bank spread)

__device__ __forceinline__ unsigned int f2bf(float f) {
    unsigned int u = __float_as_uint(f);
    unsigned int r = (u + 0x7FFFu + ((u >> 16) & 1u)) >> 16;   // RNE
    return r & 0xFFFFu;
}

__device__ __forceinline__ float fast_tanh(float x) {
    float cx = fminf(fmaxf(x, -9.f), 9.f);
    float e  = __builtin_amdgcn_exp2f(cx * 2.8853900817779268f);  // exp(2x)
    return (e - 1.f) * __builtin_amdgcn_rcpf(e + 1.f);
}

__device__ __forceinline__ v8f wmma_bf16(const ABPack& a, const ABPack& b, v8f c) {
    return __builtin_amdgcn_wmma_f32_16x16x32_bf16(
               false, a.v, false, b.v, (short)0, c, false, false);
}

// ---- kernel 1: Wc = bf16(W + U) --------------------------------------------
__global__ void prep_wc(const float* __restrict__ W, const float* __restrict__ Umat,
                        unsigned short* __restrict__ Wc) {
    int i = blockIdx.x * 256 + threadIdx.x;          // 1M elements
    Wc[i] = (unsigned short)f2bf(W[i] + Umat[i]);
}

// ---- kernel 2: fused GEMM + tanh + dot(v) + mask ---------------------------
__global__ void __launch_bounds__(256)
fused_scores(const float* __restrict__ X,
             const unsigned short* __restrict__ Wc,
             const float* __restrict__ v,
             const int* __restrict__ mask,
             float* __restrict__ scores) {
    extern __shared__ char smem_raw[];
    unsigned short* sX    = (unsigned short*)smem_raw;                 // [64][1032]
    float*          sScor = (float*)(smem_raw + MTILE * LSTRIDE * 2);  // [64]

    const int tid  = threadIdx.x;
    const int wave = tid >> 5;
    const int lane = tid & 31;
    const int rowBase = blockIdx.x * MTILE;          // global row in [B*S]

    if (tid < MTILE) sScor[tid] = 0.f;

    // ---- stage X tile (fp32 -> bf16) into LDS, 256 thr * 64 it * float4 ----
    const float* Xb = X + (size_t)rowBase * EDIM;
    for (int it = 0; it < 64; ++it) {
        int p = (it * 256 + tid) * 4;                // 0 .. 65535
        int r = p >> 10, c = p & 1023;
        float4 f = *(const float4*)(Xb + p);
        uint2 pk;
        pk.x = f2bf(f.x) | (f2bf(f.y) << 16);
        pk.y = f2bf(f.z) | (f2bf(f.w) << 16);
        *(uint2*)(sX + r * LSTRIDE + c) = pk;
    }
    __syncthreads();

    // ---- per-wave 2x2 tile assignment: 2 row-pairs x 4 N-quarters ----------
    const int rgPair = wave & 1;                     // rows 0-31 or 32-63
    const int nq     = wave >> 1;                    // 0..3 -> 256 columns
    const int rowOff  = rgPair * 32;
    const int laneMod = lane & 15;
    const int hi      = lane >> 4;                   // 0: lanes 0-15, 1: 16-31

    const unsigned short* aRow0 = sX + (rowOff + laneMod) * LSTRIDE;
    const unsigned short* aRow1 = aRow0 + 16 * LSTRIDE;

    float sacc0[8], sacc1[8];
#pragma unroll
    for (int r = 0; r < 8; ++r) { sacc0[r] = 0.f; sacc1[r] = 0.f; }

#pragma clang loop unroll(disable)
    for (int nt = 0; nt < 8; ++nt) {
        const int n0 = nq * 256 + nt * 32;           // two n-tiles: n0, n0+16
        const unsigned short* bp0 = Wc + (size_t)(n0 + laneMod) * EDIM + hi * 8;
        const unsigned short* bp1 = bp0 + (size_t)16 * EDIM;

        // Opaque zero: a fresh unknown value per nt iteration. Prevents LICM
        // from hoisting the nt-invariant LDS A loads out of this loop (which
        // would spill ~2KB/lane to scratch), while keeping ds_load lowering.
        int koff = 0;
        asm volatile("" : "+v"(koff));

        v8f acc00 = {}, acc01 = {}, acc10 = {}, acc11 = {};

        // A layout: lanes 0-15 hold K {k0..k0+7, k0+16..k0+23};
        //           lanes 16-31 hold K {k0+8..k0+15, k0+24..k0+31}
        // B layout: same K order as A
#pragma unroll 4
        for (int kc = 0; kc < 32; ++kc) {
            const int k0 = kc * 32;
            const int kA = k0 + hi * 8 + koff;
            ABPack a0, a1, b0, b1;
            a0.q[0] = *(const v4u*)(aRow0 + kA);
            a0.q[1] = *(const v4u*)(aRow0 + kA + 16);
            a1.q[0] = *(const v4u*)(aRow1 + kA);
            a1.q[1] = *(const v4u*)(aRow1 + kA + 16);
            b0.q[0] = *(const v4u*)(bp0 + k0);
            b0.q[1] = *(const v4u*)(bp0 + k0 + 16);
            b1.q[0] = *(const v4u*)(bp1 + k0);
            b1.q[1] = *(const v4u*)(bp1 + k0 + 16);

            acc00 = wmma_bf16(a0, b0, acc00);
            acc01 = wmma_bf16(a0, b1, acc01);
            acc10 = wmma_bf16(a1, b0, acc10);
            acc11 = wmma_bf16(a1, b1, acc11);
        }

        // epilogue: tanh + weight by v[n]; C layout: VGPR r = row base+r
        // (lanes 0-15) / base+r+8 (lanes 16-31), N = lane&15
        const float vv0 = v[n0 + laneMod];
        const float vv1 = v[n0 + 16 + laneMod];
#pragma unroll
        for (int r = 0; r < 8; ++r) {
            sacc0[r] += fast_tanh(acc00[r]) * vv0 + fast_tanh(acc01[r]) * vv1;
            sacc1[r] += fast_tanh(acc10[r]) * vv0 + fast_tanh(acc11[r]) * vv1;
        }
    }

    // ---- cross-lane reduce within each 16-lane half, accumulate in LDS -----
#pragma unroll
    for (int r = 0; r < 8; ++r) {
        float s0 = sacc0[r];
        s0 += __shfl_xor(s0, 1, 32);
        s0 += __shfl_xor(s0, 2, 32);
        s0 += __shfl_xor(s0, 4, 32);
        s0 += __shfl_xor(s0, 8, 32);
        float s1 = sacc1[r];
        s1 += __shfl_xor(s1, 1, 32);
        s1 += __shfl_xor(s1, 2, 32);
        s1 += __shfl_xor(s1, 4, 32);
        s1 += __shfl_xor(s1, 8, 32);
        if (lane == 0) {
            atomicAdd(&sScor[rowOff + r],          s0);
            atomicAdd(&sScor[rowOff + 16 + r],     s1);
        }
        if (lane == 16) {
            atomicAdd(&sScor[rowOff + r + 8],      s0);
            atomicAdd(&sScor[rowOff + 16 + r + 8], s1);
        }
    }
    __syncthreads();

    if (tid < MTILE) {
        int g = rowBase + tid;
        scores[g] = (mask[g] != 0) ? sScor[tid] : -INFINITY;
    }
}

// ---- kernel 3: masked softmax over S (in place) ----------------------------
__global__ void softmax_rows(float* __restrict__ scores) {
    __shared__ float red[256];
    float* row = scores + (size_t)blockIdx.x * SEQ;
    const int tid = threadIdx.x;

    float vals[8];
    float m = -INFINITY;
#pragma unroll
    for (int i = 0; i < 8; ++i) { vals[i] = row[tid + i * 256]; m = fmaxf(m, vals[i]); }
    red[tid] = m; __syncthreads();
    for (int s = 128; s > 0; s >>= 1) {
        if (tid < s) red[tid] = fmaxf(red[tid], red[tid + s]);
        __syncthreads();
    }
    m = red[0]; __syncthreads();

    float sum = 0.f;
#pragma unroll
    for (int i = 0; i < 8; ++i) {
        float e = (vals[i] == -INFINITY)
                      ? 0.f
                      : __builtin_amdgcn_exp2f((vals[i] - m) * 1.4426950408889634f);
        vals[i] = e; sum += e;
    }
    red[tid] = sum; __syncthreads();
    for (int s = 128; s > 0; s >>= 1) {
        if (tid < s) red[tid] += red[tid + s];
        __syncthreads();
    }
    float inv = __builtin_amdgcn_rcpf(red[0]);
#pragma unroll
    for (int i = 0; i < 8; ++i) row[tid + i * 256] = vals[i] * inv;
}

// ---- kernel 4: out[b,e] = sum_s alpha[b,s] * X[b,s,e] ----------------------
__global__ void weighted_sum(const float* __restrict__ X,
                             const float* __restrict__ alpha,
                             float* __restrict__ out) {
    const int b = blockIdx.y;
    const int e = blockIdx.x * 256 + threadIdx.x;
    const float* Xb = X + (size_t)b * SEQ * EDIM + e;
    const float* ab = alpha + (size_t)b * SEQ;
    float acc = 0.f;
#pragma unroll 8
    for (int s = 0; s < SEQ; ++s)
        acc += ab[s] * Xb[(size_t)s * EDIM];
    out[(size_t)b * EDIM + e] = acc;
}

// ---------------------------------------------------------------------------
extern "C" void kernel_launch(void* const* d_in, const int* in_sizes, int n_in,
                              void* d_out, int out_size, void* d_ws, size_t ws_size,
                              hipStream_t stream) {
    const float* X    = (const float*)d_in[0];   // [B,S,E]
    const int*   mask = (const int*)d_in[1];     // [B,S]
    const float* W    = (const float*)d_in[2];   // [U,E]
    const float* Umat = (const float*)d_in[3];   // [U,E]
    const float* v    = (const float*)d_in[4];   // [1,U]
    float* out = (float*)d_out;                  // [B,E]

    unsigned short* Wc = (unsigned short*)d_ws;                      // 2 MB bf16
    float* scores = (float*)((char*)d_ws + (size_t)UDIM * EDIM * 2); // 256 KB

    prep_wc<<<(UDIM * EDIM) / 256, 256, 0, stream>>>(W, Umat, Wc);

    const size_t smem = (size_t)MTILE * LSTRIDE * 2 + MTILE * 4;     // ~132 KB
    hipFuncSetAttribute(reinterpret_cast<const void*>(fused_scores),
                        hipFuncAttributeMaxDynamicSharedMemorySize, (int)smem);
    fused_scores<<<(BATCH * SEQ) / MTILE, 256, smem, stream>>>(X, Wc, v, mask, scores);

    softmax_rows<<<BATCH, 256, 0, stream>>>(scores);

    weighted_sum<<<dim3(EDIM / 256, BATCH), 256, 0, stream>>>(X, scores, out);
}
